// DTITAG_4810363372766
// MI455X (gfx1250) — matrix operand order repaired and checked by reading, not compiled, not run on hardware
//
#include <hip/hip_runtime.h>
#include <hip/hip_bf16.h>
#include <math.h>

// ---------------------------------------------------------------------------
// Problem constants (match reference)
// ---------------------------------------------------------------------------
#define NNODES 160000
#define NEDGES 1280000
#define NGRAPH 800
#define NPG    200
#define HID    64
#define INF_   31

typedef __attribute__((ext_vector_type(16))) _Float16 v16h;
typedef __attribute__((ext_vector_type(8)))  float    v8f;

// ---------------------------------------------------------------------------
// Degree / norm
// ---------------------------------------------------------------------------
__global__ __launch_bounds__(256) void count_deg_kernel(
    const int* __restrict__ dst, float* __restrict__ deg, int nE)
{
    int e = blockIdx.x * blockDim.x + threadIdx.x;
    if (e < nE) unsafeAtomicAdd(&deg[dst[e]], 1.0f);
}

__global__ __launch_bounds__(256) void norm_kernel(
    const float* __restrict__ deg, float* __restrict__ norm, int n)
{
    int i = blockIdx.x * blockDim.x + threadIdx.x;
    if (i < n) {
        float d = deg[i];
        d = d < 1.0f ? 1.0f : d;
        norm[i] = rsqrtf(d);
    }
}

// ---------------------------------------------------------------------------
// Edge scatter: one wave (32 lanes) per edge. hout[dst] += hin[src]*norm[src]
// 32 lanes read consecutive floats -> one coalesced line per edge; atomics
// land in L2 (h fits in 192MB L2).
// ---------------------------------------------------------------------------
__global__ __launch_bounds__(256) void scatter_edges_kernel(
    const float* __restrict__ hin, const float* __restrict__ norm,
    const int* __restrict__ src, const int* __restrict__ dst,
    float* __restrict__ hout, int F, int nE)
{
    int gid  = blockIdx.x * blockDim.x + threadIdx.x;
    int e    = gid >> 5;
    int lane = gid & 31;
    if (e >= nE) return;
    int s = src[e];
    int d = dst[e];
    float ns = norm[s];
    const float* hs = hin  + (size_t)s * F;
    float*       hd = hout + (size_t)d * F;
    for (int f = lane; f < F; f += 32)
        unsafeAtomicAdd(&hd[f], hs[f] * ns);
}

// hout[i][f] *= norm[i]   (dst-side symmetric normalization)
__global__ __launch_bounds__(256) void scale_dst_kernel(
    float* __restrict__ hout, const float* __restrict__ norm, int F, int total)
{
    int idx = blockIdx.x * blockDim.x + threadIdx.x;
    if (idx >= total) return;
    int node = idx / F;
    hout[idx] *= norm[node];
}

// ---------------------------------------------------------------------------
// TAGConv linear:  out[N,64] = relu( concat(h0,h1,h2)[N,3F] @ W[3F,64] + b )
// WMMA f16 16x16x32, f32 accumulate. 128 threads = 4 wave32; each wave owns a
// 16x16 tile of the 16x64 output block. Persistent blocks loop over row tiles
// so the W->LDS staging cost is amortized.
// ---------------------------------------------------------------------------
#define KMAX 192

__global__ __launch_bounds__(128) void tag_gemm_kernel(
    const float* __restrict__ p0, const float* __restrict__ p1,
    const float* __restrict__ p2, int F, int Kreal, int Kpad,
    const float* __restrict__ W, const float* __restrict__ bias,
    float* __restrict__ out, int nrb)
{
    __shared__ _Float16 Wlds[HID * KMAX];   // transposed: Wlds[n*Kpad + k]
    __shared__ _Float16 Alds[16 * KMAX];    // Alds[m*Kpad + k]

    const int tid  = threadIdx.x;
    const int lane = tid & 31;
    const int wave = tid >> 5;

    // Stage W (fp32 -> f16), transposed so each lane's B fragment is 16
    // contiguous halves (two ds_load_b128). Pad K region zero-filled.
    for (int idx = tid; idx < HID * Kpad; idx += 128) {
        int k = idx >> 6;            // global read idx = k*64 + n (coalesced)
        int n = idx & 63;
        float w = (k < Kreal) ? W[idx] : 0.0f;
        Wlds[n * Kpad + k] = (_Float16)w;
    }
    __syncthreads();

    const int m      = lane & 15;
    const int hi     = lane >> 4;     // 0: lanes 0-15, 1: lanes 16-31
    const int kbaseA = hi * 16;       // A lanes 16-31 hold K=16..31
    const int kbaseB = hi * 16;       // B lanes 16-31 hold K=16..31
    const int n      = wave * 16 + m;
    const int nchunks = Kpad >> 5;

    for (int rb = blockIdx.x; rb < nrb; rb += gridDim.x) {
        const int row0 = rb * 16;

        // Stage A tile: 16 rows x Kpad, gathered from the three hop buffers.
        for (int idx = tid; idx < 16 * Kpad; idx += 128) {
            int r = idx / Kpad;
            int k = idx - r * Kpad;
            size_t node = (size_t)(row0 + r);
            float v;
            if      (k < F)       v = p0[node * F + k];
            else if (k < 2 * F)   v = p1[node * F + (k - F)];
            else if (k < Kreal)   v = p2[node * F + (k - 2 * F)];
            else                  v = 0.0f;
            Alds[idx] = (_Float16)v;
        }
        __syncthreads();

        v8f acc = {};
        for (int c = 0; c < nchunks; ++c) {
            const int k0 = c * 32;
            // A fragment: 16 contiguous K (same order as B).
            union { v16h v; unsigned int u[8]; } a;
            const unsigned int* arow =
                (const unsigned int*)(&Alds[m * Kpad + k0 + kbaseA]);
            a.u[0] = arow[0];  a.u[1] = arow[1];
            a.u[2] = arow[2];  a.u[3] = arow[3];
            a.u[4] = arow[4];  a.u[5] = arow[5];
            a.u[6] = arow[6]; a.u[7] = arow[7];
            // B fragment: 16 contiguous halves (32B aligned: Kpad%16==0).
            v16h b = *(const v16h*)(&Wlds[n * Kpad + k0 + kbaseB]);
            acc = __builtin_amdgcn_wmma_f32_16x16x32_f16(
                false, a.v, false, b, (short)0, acc, false, false);
        }
        __syncthreads();   // all waves done reading Alds before next staging

        // Epilogue: bias + ReLU. C layout: VGPR r -> M = r + 8*hi, N = n.
        float bn = bias[n];
        #pragma unroll
        for (int r = 0; r < 8; ++r) {
            int row = row0 + r + hi * 8;
            float v = acc[r] + bn;
            out[(size_t)row * HID + n] = v > 0.0f ? v : 0.0f;
        }
    }
}

// ---------------------------------------------------------------------------
// gate[i] = h[i,:] . Wg + bg   (wave per node, shuffle reduce)
// ---------------------------------------------------------------------------
__global__ __launch_bounds__(256) void gate_kernel(
    const float* __restrict__ h, const float* __restrict__ Wg,
    const float* __restrict__ bg, float* __restrict__ gate, int n)
{
    int gid  = blockIdx.x * blockDim.x + threadIdx.x;
    int node = gid >> 5;
    int lane = gid & 31;
    if (node >= n) return;
    const float* hr = h + (size_t)node * HID;
    float s = hr[lane] * Wg[lane] + hr[lane + 32] * Wg[lane + 32];
    for (int o = 16; o > 0; o >>= 1) s += __shfl_xor(s, o, 32);
    if (lane == 0) gate[node] = s + bg[0];
}

// ---------------------------------------------------------------------------
// Global attention pooling: one block per graph (nodes contiguous, 200/graph)
// ---------------------------------------------------------------------------
__global__ __launch_bounds__(256) void pool_kernel(
    const float* __restrict__ h, const float* __restrict__ gate,
    float* __restrict__ out)
{
    __shared__ float ebuf[NPG];
    __shared__ float red[256];
    const int g    = blockIdx.x;
    const int t    = threadIdx.x;
    const int base = g * NPG;

    float gv = (t < NPG) ? gate[base + t] : -3.402823466e38f;
    red[t] = gv;
    __syncthreads();
    for (int o = 128; o > 0; o >>= 1) {
        if (t < o) red[t] = fmaxf(red[t], red[t + o]);
        __syncthreads();
    }
    float gmax = red[0];
    __syncthreads();

    float ev = (t < NPG) ? expf(gv - gmax) : 0.0f;
    if (t < NPG) ebuf[t] = ev;
    red[t] = ev;
    __syncthreads();
    for (int o = 128; o > 0; o >>= 1) {
        if (t < o) red[t] += red[t + o];
        __syncthreads();
    }
    float z = red[0];
    __syncthreads();

    if (t < HID) {
        float acc = 0.0f;
        for (int nl = 0; nl < NPG; ++nl)
            acc += ebuf[nl] * h[(size_t)(base + nl) * HID + t];  // coalesced
        out[g * HID + t] = acc / z;
    }
}

// ---------------------------------------------------------------------------
// Host orchestration
// ---------------------------------------------------------------------------
extern "C" void kernel_launch(void* const* d_in, const int* in_sizes, int n_in,
                              void* d_out, int out_size, void* d_ws, size_t ws_size,
                              hipStream_t stream)
{
    const float* x   = (const float*)d_in[0];
    const int*   src = (const int*)d_in[1];
    const int*   dst = (const int*)d_in[2];
    // d_in[3] node2graph: unused (graphs are contiguous, NPG nodes each)
    const float* Wl[4] = { (const float*)d_in[4], (const float*)d_in[6],
                           (const float*)d_in[8], (const float*)d_in[10] };
    const float* bl[4] = { (const float*)d_in[5], (const float*)d_in[7],
                           (const float*)d_in[9], (const float*)d_in[11] };
    const float* Wg = (const float*)d_in[12];
    const float* bg = (const float*)d_in[13];

    // Workspace layout (floats)
    float* ws   = (float*)d_ws;
    float* norm = ws;                                  // N
    float* deg  = norm + NNODES;                       // N
    float* hop1 = deg  + NNODES;                       // N*64
    float* hop2 = hop1 + (size_t)NNODES * HID;         // N*64
    float* hA   = hop2 + (size_t)NNODES * HID;         // N*64
    float* hB   = hA   + (size_t)NNODES * HID;         // N*64
    float* gate = hB   + (size_t)NNODES * HID;         // N

    // Degree + symmetric norm
    hipMemsetAsync(deg, 0, (size_t)NNODES * sizeof(float), stream);
    count_deg_kernel<<<NEDGES / 256, 256, 0, stream>>>(dst, deg, NEDGES);
    norm_kernel<<<(NNODES + 255) / 256, 256, 0, stream>>>(deg, norm, NNODES);

    const int scatter_blocks = (NEDGES * 32) / 256;    // 1 wave/edge

    auto run_layer = [&](const float* h_in, int F, const float* W,
                         const float* b, float* h_out) {
        const int Kreal = 3 * F;
        const int Kpad  = (Kreal + 31) / 32 * 32;
        const size_t bytes = (size_t)NNODES * F * sizeof(float);
        const int total = NNODES * F;

        hipMemsetAsync(hop1, 0, bytes, stream);
        scatter_edges_kernel<<<scatter_blocks, 256, 0, stream>>>(
            h_in, norm, src, dst, hop1, F, NEDGES);
        scale_dst_kernel<<<(total + 255) / 256, 256, 0, stream>>>(
            hop1, norm, F, total);

        hipMemsetAsync(hop2, 0, bytes, stream);
        scatter_edges_kernel<<<scatter_blocks, 256, 0, stream>>>(
            hop1, norm, src, dst, hop2, F, NEDGES);
        scale_dst_kernel<<<(total + 255) / 256, 256, 0, stream>>>(
            hop2, norm, F, total);

        tag_gemm_kernel<<<2048, 128, 0, stream>>>(
            h_in, hop1, hop2, F, Kreal, Kpad, W, b, h_out, NNODES / 16);
    };

    run_layer(x,  INF_, Wl[0], bl[0], hA);   // K = 93 -> pad 96
    run_layer(hA, HID,  Wl[1], bl[1], hB);   // K = 192
    run_layer(hB, HID,  Wl[2], bl[2], hA);
    run_layer(hA, HID,  Wl[3], bl[3], hB);

    gate_kernel<<<(NNODES * 32) / 256, 256, 0, stream>>>(hB, Wg, bg, gate, NNODES);
    pool_kernel<<<NGRAPH, 256, 0, stream>>>(hB, gate, (float*)d_out);
}
